// SA2M_15212774162984
// MI455X (gfx1250) — matrix operand order repaired and checked by reading, not compiled, not run on hardware
//
#include <hip/hip_runtime.h>
#include <hip/hip_bf16.h>
#include <math.h>

// ---------------------------------------------------------------------------
// SA2M on MI455X (gfx1250, wave32).
// B=16, C=512, H=W=64, CI=64.  GEMMs via v_wmma_f32_16x16x32_bf16.
// Block tile: 32(M) x 128(N), K-tile 64; 8 waves, 4 WMMA / wave / K-tile.
// ---------------------------------------------------------------------------

typedef __attribute__((ext_vector_type(16))) __bf16 v16bf;
typedef __attribute__((ext_vector_type(8)))  __bf16 v8bf;
typedef __attribute__((ext_vector_type(4)))  __bf16 v4bf;
typedef __attribute__((ext_vector_type(8)))  float  v8f;

#define HW    4096
#define CCH   512
#define BB    16
#define EPSV  1e-5f
#define LDST  72      // LDS row stride in bf16 (144B = 9*16B, bank-rotating)

// ---------------- WMMA helpers (ISA 7.12.2 layouts) ------------------------

__device__ __forceinline__ v8f wmma_bf16(v16bf a, v16bf b, v8f c) {
  return __builtin_amdgcn_wmma_f32_16x16x32_bf16(false, a, false, b,
                                                 (short)0, c, false, false);
}

// A fragment: 16x32 bf16 at (rowBase, colBase) in an LDS tile of stride LDST.
// lane<16: row=lane, K {0..7,16..23}; lane>=16: row=lane-16, K {8..15,24..31}.
__device__ __forceinline__ v16bf load_frag_A(const __bf16* tile, int lane) {
  int m = lane & 15;
  int koff = (lane >> 4) << 3;            // 0 or 8
  const __bf16* row = tile + m * LDST;
  v8bf lo = *reinterpret_cast<const v8bf*>(row + koff);
  v8bf hi = *reinterpret_cast<const v8bf*>(row + 16 + koff);
  return __builtin_shufflevector(lo, hi, 0,1,2,3,4,5,6,7,8,9,10,11,12,13,14,15);
}

// B fragment: 32x16 bf16, stored transposed Bs[col][k].
// lane<16: col=lane, K {0..7,16..23}; lane>=16: col=lane-16, K {8..15,24..31}.
__device__ __forceinline__ v16bf load_frag_B(const __bf16* tile, int lane) {
  int n  = lane & 15;
  int kh = (lane >> 4) << 3;              // 0 or 8
  const __bf16* row = tile + n * LDST;
  v8bf lo = *reinterpret_cast<const v8bf*>(row + kh);
  v8bf hi = *reinterpret_cast<const v8bf*>(row + 16 + kh);
  return __builtin_shufflevector(lo, hi, 0,1,2,3,4,5,6,7,8,9,10,11,12,13,14,15);
}

// ---------------- conv1x1 as GEMM:  Y(OxN) = W(OxK) @ X(KxN) per batch -----
// fp32 inputs -> bf16 LDS tiles -> fp32-accum WMMA.
// grid(N/128, O/32, B), block 256.
template <bool OUTBF>
__global__ void k_conv1x1_wmma(const float* __restrict__ W,
                               const float* __restrict__ X,
                               const float* __restrict__ bias,
                               float* __restrict__ outF,
                               __bf16* __restrict__ outB,
                               int O, int K, int N) {
  alignas(32) __shared__ __bf16 As[32][LDST];
  alignas(32) __shared__ __bf16 Bs[128][LDST];

  const int tid  = threadIdx.x;
  const int lane = tid & 31;
  const int wv   = tid >> 5;
  const int MBase = blockIdx.y * 32;
  const int NBase = blockIdx.x * 128;

  X += (size_t)blockIdx.z * K * N;

  v8f acc0 = {0.f,0.f,0.f,0.f,0.f,0.f,0.f,0.f};
  v8f acc1 = acc0;

  const int am = tid >> 3, ak = (tid & 7) * 8;   // A copy: 32 rows x 64 k
  const int bk = tid >> 5, bn = (tid & 31) * 4;  // B copy: 64 rows x 128 n

  for (int k0 = 0; k0 < K; k0 += 64) {
    // stage global loads in registers first (batched under one loadcnt wait)
    const float* aw = &W[(MBase + am) * K + k0 + ak];
    float4 a0 = *(const float4*)aw;
    float4 a1 = *(const float4*)(aw + 4);
    float4 bv[8];
#pragma unroll
    for (int p = 0; p < 8; p++)
      bv[p] = *(const float4*)&X[(k0 + bk + p * 8) * N + NBase + bn];

    __syncthreads();
    {
      __bf16* d = &As[am][ak];
      d[0]=(__bf16)a0.x; d[1]=(__bf16)a0.y; d[2]=(__bf16)a0.z; d[3]=(__bf16)a0.w;
      d[4]=(__bf16)a1.x; d[5]=(__bf16)a1.y; d[6]=(__bf16)a1.z; d[7]=(__bf16)a1.w;
    }
#pragma unroll
    for (int p = 0; p < 8; p++) {
      int k = bk + p * 8;
      Bs[bn+0][k]=(__bf16)bv[p].x; Bs[bn+1][k]=(__bf16)bv[p].y;
      Bs[bn+2][k]=(__bf16)bv[p].z; Bs[bn+3][k]=(__bf16)bv[p].w;
    }
    __syncthreads();

#pragma unroll
    for (int kk = 0; kk < 64; kk += 32) {
      v16bf b  = load_frag_B(&Bs[wv * 16][kk], lane);
      v16bf a0f = load_frag_A(&As[0][kk],  lane);
      v16bf a1f = load_frag_A(&As[16][kk], lane);
      acc0 = wmma_bf16(a0f, b, acc0);
      acc1 = wmma_bf16(a1f, b, acc1);
    }
  }

  const int n   = NBase + wv * 16 + (lane & 15);
  const int mhi = (lane >> 4) * 8;
  const size_t cb = (size_t)blockIdx.z * O * N;
#pragma unroll
  for (int i = 0; i < 8; i++) {
    int m0 = MBase + mhi + i;
    int m1 = m0 + 16;
    float v0 = acc0[i] + (bias ? bias[m0] : 0.f);
    float v1 = acc1[i] + (bias ? bias[m1] : 0.f);
    if constexpr (OUTBF) {
      outB[cb + (size_t)m0 * N + n] = (__bf16)v0;
      outB[cb + (size_t)m1 * N + n] = (__bf16)v1;
    } else {
      outF[cb + (size_t)m0 * N + n] = v0;
      outF[cb + (size_t)m1 * N + n] = v1;
    }
  }
}

// ---------------- batched 512x512x512 bf16 GEMM (attention views) ----------
// D(MxN) = Aeff(MxK) @ B(KxN); TRANSA: Aeff[m][k] = A[k*lda + m].
// grid(N/128, M/32, B), block 256.
template <bool TRANSA, bool OUTBF>
__global__ void k_gemm_bf16(const __bf16* __restrict__ A, int lda, size_t abatch,
                            const __bf16* __restrict__ Bm, int ldb, size_t bbatch,
                            float* __restrict__ outF, __bf16* __restrict__ outB,
                            int ldc, size_t cbatch, int K) {
  alignas(32) __shared__ __bf16 As[32][LDST];
  alignas(32) __shared__ __bf16 Bs[128][LDST];

  const int tid  = threadIdx.x;
  const int lane = tid & 31;
  const int wv   = tid >> 5;
  const int MBase = blockIdx.y * 32;
  const int NBase = blockIdx.x * 128;

  A  += (size_t)blockIdx.z * abatch;
  Bm += (size_t)blockIdx.z * bbatch;

  v8f acc0 = {0.f,0.f,0.f,0.f,0.f,0.f,0.f,0.f};
  v8f acc1 = acc0;

  const int am = tid >> 3, ak = (tid & 7) * 8;   // !TRANSA copy map
  const int tk = tid >> 3, tm = (tid & 7) * 4;   // TRANSA copy map
  const int bk = tid >> 5, bn = (tid & 31) * 4;  // B copy map

  for (int k0 = 0; k0 < K; k0 += 64) {
    // register staging
    v8bf aReg;            // !TRANSA: 8 contiguous k
    v4bf aRegT[2];        // TRANSA: 4 contiguous m, two k-passes
    if constexpr (!TRANSA) {
      aReg = *(const v8bf*)&A[(MBase + am) * lda + k0 + ak];
    } else {
#pragma unroll
      for (int p = 0; p < 2; p++)
        aRegT[p] = *(const v4bf*)&A[(k0 + tk + p * 32) * lda + MBase + tm];
    }
    v4bf bReg[8];
#pragma unroll
    for (int p = 0; p < 8; p++)
      bReg[p] = *(const v4bf*)&Bm[(k0 + bk + p * 8) * ldb + NBase + bn];

    __syncthreads();
    if constexpr (!TRANSA) {
      *(v8bf*)&As[am][ak] = aReg;
    } else {
#pragma unroll
      for (int p = 0; p < 2; p++) {
        int k = tk + p * 32;
        As[tm+0][k]=aRegT[p].x; As[tm+1][k]=aRegT[p].y;
        As[tm+2][k]=aRegT[p].z; As[tm+3][k]=aRegT[p].w;
      }
    }
#pragma unroll
    for (int p = 0; p < 8; p++) {
      int k = bk + p * 8;
      Bs[bn+0][k]=bReg[p].x; Bs[bn+1][k]=bReg[p].y;
      Bs[bn+2][k]=bReg[p].z; Bs[bn+3][k]=bReg[p].w;
    }
    __syncthreads();

#pragma unroll
    for (int kk = 0; kk < 64; kk += 32) {
      v16bf b  = load_frag_B(&Bs[wv * 16][kk], lane);
      v16bf a0f = load_frag_A(&As[0][kk],  lane);
      v16bf a1f = load_frag_A(&As[16][kk], lane);
      acc0 = wmma_bf16(a0f, b, acc0);
      acc1 = wmma_bf16(a1f, b, acc1);
    }
  }

  const int n   = NBase + wv * 16 + (lane & 15);
  const int mhi = (lane >> 4) * 8;
  const size_t cb = (size_t)blockIdx.z * cbatch;
#pragma unroll
  for (int i = 0; i < 8; i++) {
    int m0 = MBase + mhi + i;
    int m1 = m0 + 16;
    if constexpr (OUTBF) {
      outB[cb + (size_t)m0 * ldc + n] = (__bf16)acc0[i];
      outB[cb + (size_t)m1 * ldc + n] = (__bf16)acc1[i];
    } else {
      outF[cb + (size_t)m0 * ldc + n] = acc0[i];
      outF[cb + (size_t)m1 * ldc + n] = acc1[i];
    }
  }
}

// ---------------- reductions / small kernels -------------------------------

__global__ void k_row_mean_f32(const float* __restrict__ src,
                               float* __restrict__ dst, int len) {
  __shared__ float red[256];
  const float* p = src + (size_t)blockIdx.x * len;
  float s = 0.f;
  for (int i = threadIdx.x; i < len; i += 256) s += p[i];
  red[threadIdx.x] = s; __syncthreads();
  for (int st = 128; st > 0; st >>= 1) {
    if (threadIdx.x < st) red[threadIdx.x] += red[threadIdx.x + st];
    __syncthreads();
  }
  if (threadIdx.x == 0) dst[blockIdx.x] = red[0] / (float)len;
}

__global__ void k_row_mean_bf16(const __bf16* __restrict__ src,
                                float* __restrict__ dst, int len) {
  __shared__ float red[256];
  const __bf16* p = src + (size_t)blockIdx.x * len;
  float s = 0.f;
  for (int i = threadIdx.x; i < len; i += 256) s += (float)p[i];
  red[threadIdx.x] = s; __syncthreads();
  for (int st = 128; st > 0; st >>= 1) {
    if (threadIdx.x < st) red[threadIdx.x] += red[threadIdx.x + st];
    __syncthreads();
  }
  if (threadIdx.x == 0) dst[blockIdx.x] = red[0] / (float)len;
}

// AC branch: ap(16x512) -> fc(512->32) -> fc(32->512) -> BN(axis=0) -> relu
__global__ void k_ac(const float* __restrict__ ap,
                     const float* __restrict__ w0, const float* __restrict__ b0,
                     const float* __restrict__ w1, const float* __restrict__ b1,
                     const float* __restrict__ gamma, const float* __restrict__ beta,
                     float* __restrict__ acout) {
  __shared__ float t0s[16][32];
  __shared__ float t1s[16][512];
  const int tid = threadIdx.x;
  {
    int b = tid >> 5, o = tid & 31;
    float s = b0[o];
    const float* ar = ap + b * 512;
    const float* wr = w0 + o * 512;
    for (int c = 0; c < 512; c++) s = fmaf(ar[c], wr[c], s);
    t0s[b][o] = s;
  }
  __syncthreads();
  {
    int c = tid;
    const float* wr = w1 + c * 32;
    for (int b = 0; b < 16; b++) {
      float s = b1[c];
      for (int o = 0; o < 32; o++) s = fmaf(t0s[b][o], wr[o], s);
      t1s[b][c] = s;
    }
    float mu = 0.f;
    for (int b = 0; b < 16; b++) mu += t1s[b][c];
    mu *= (1.f / 16.f);
    float var = 0.f;
    for (int b = 0; b < 16; b++) { float d = t1s[b][c] - mu; var += d * d; }
    var *= (1.f / 16.f);
    float rs = rsqrtf(var + EPSV) * gamma[c];
    for (int b = 0; b < 16; b++) {
      float v = (t1s[b][c] - mu) * rs + beta[c];
      acout[b * 512 + c] = fmaxf(v, 0.f);
    }
  }
}

// 3x3 conv 32->32 ch, pad 1.  grid(16, 32, B)
__global__ void k_conv3x3(const float* __restrict__ y1,
                          const float* __restrict__ w3,
                          const float* __restrict__ b3,
                          float* __restrict__ y3) {
  __shared__ float ws[288];
  const int b = blockIdx.z, oc = blockIdx.y;
  const int p = blockIdx.x * 256 + threadIdx.x;
  for (int i = threadIdx.x; i < 288; i += 256) ws[i] = w3[oc * 288 + i];
  __syncthreads();
  const int yy = p >> 6, xx = p & 63;
  float acc = b3[oc];
  const float* src = y1 + (size_t)b * 32 * HW;
  for (int ic = 0; ic < 32; ic++) {
    const float* s = src + ic * HW;
    const float* w = ws + ic * 9;
    for (int r = 0; r < 3; r++) {
      int iy = yy + r - 1;
      if (iy < 0 || iy > 63) continue;
      for (int c = 0; c < 3; c++) {
        int ix = xx + c - 1;
        if (ix < 0 || ix > 63) continue;
        acc = fmaf(w[r * 3 + c], s[iy * 64 + ix], acc);
      }
    }
  }
  y3[((size_t)b * 32 + oc) * HW + p] = acc;
}

// y = relu(w2 @ y3 + b2); accumulate per-(b,c) sum(y), sum(y^2) on the fly.
__global__ void k_als2_reduce(const float* __restrict__ y3,
                              const float* __restrict__ w2,
                              const float* __restrict__ b2,
                              float* __restrict__ s1, float* __restrict__ s2) {
  const int b = blockIdx.y;
  const int p = blockIdx.x * 256 + threadIdx.x;
  const int lane = threadIdx.x & 31;
  float r[32];
  const float* base = y3 + (size_t)b * 32 * HW + p;
  for (int ic = 0; ic < 32; ic++) r[ic] = base[ic * HW];
  for (int oc = 0; oc < 512; oc++) {
    const float* w = w2 + oc * 32;
    float v = b2[oc];
    for (int ic = 0; ic < 32; ic++) v = fmaf(w[ic], r[ic], v);
    v = fmaxf(v, 0.f);
    float q = v * v;
    for (int o = 16; o; o >>= 1) { v += __shfl_xor(v, o); q += __shfl_xor(q, o); }
    if (lane == 0) {
      atomicAdd(&s1[b * 512 + oc], v);
      atomicAdd(&s2[b * 512 + oc], q);
    }
  }
}

// gap_als[b,c] = (mean_hw(y) - mu_c) * rsqrt(var_c + eps) * gamma + beta
__global__ void k_gap_als(const float* __restrict__ s1, const float* __restrict__ s2,
                          const float* __restrict__ gamma, const float* __restrict__ beta,
                          float* __restrict__ g) {
  const int c = threadIdx.x;
  float a1 = 0.f, a2 = 0.f;
  for (int b = 0; b < 16; b++) { a1 += s1[b * 512 + c]; a2 += s2[b * 512 + c]; }
  const float inv = 1.f / (16.f * (float)HW);
  float mu = a1 * inv;
  float var = a2 * inv - mu * mu;
  float rs = rsqrtf(var + EPSV) * gamma[c];
  for (int b = 0; b < 16; b++)
    g[b * 512 + c] = (s1[b * 512 + c] * (1.f / (float)HW) - mu) * rs + beta[c];
}

// row-wise softmax of AttnT (rows = m, contiguous n), write bf16
__global__ void k_softmax_rows(const float* __restrict__ A, __bf16* __restrict__ Ob) {
  __shared__ float red[256];
  const size_t row = (size_t)blockIdx.y * 512 + blockIdx.x;
  const float* src = A + row * 512;
  __bf16* dst = Ob + row * 512;
  const int tid = threadIdx.x;
  float v0 = src[tid], v1 = src[tid + 256];
  red[tid] = fmaxf(v0, v1); __syncthreads();
  for (int s = 128; s > 0; s >>= 1) {
    if (tid < s) red[tid] = fmaxf(red[tid], red[tid + s]);
    __syncthreads();
  }
  float mx = red[0]; __syncthreads();
  float e0 = __expf(v0 - mx), e1 = __expf(v1 - mx);
  red[tid] = e0 + e1; __syncthreads();
  for (int s = 128; s > 0; s >>= 1) {
    if (tid < s) red[tid] += red[tid + s];
    __syncthreads();
  }
  float inv = 1.f / red[0];
  dst[tid] = (__bf16)(e0 * inv);
  dst[tid + 256] = (__bf16)(e1 * inv);
}

// gap_ags[b,co] = mask_w[co,:] @ out_mean[b,:] + ap[b,co]
__global__ void k_gap_ags(const float* __restrict__ ap,
                          const float* __restrict__ mw,
                          const float* __restrict__ om,
                          float* __restrict__ g) {
  const int co = threadIdx.x;
  const float* w = mw + co * 64;
  for (int b = 0; b < 16; b++) {
    float acc = ap[b * 512 + co];
    const float* o = om + b * 64;
    for (int ci = 0; ci < 64; ci++) acc = fmaf(w[ci], o[ci], acc);
    g[b * 512 + co] = acc;
  }
}

// m = max3 -> channel mean -> batch softmax (16 scalars)
__global__ void k_final_sm(const float* __restrict__ ac, const float* __restrict__ gals,
                           const float* __restrict__ gags, float* __restrict__ sm) {
  __shared__ float mr[16];
  const int tid = threadIdx.x;
  const int b = tid >> 5, lane = tid & 31;
  float s = 0.f;
  for (int c = lane; c < 512; c += 32) {
    int i = b * 512 + c;
    s += fmaxf(fmaxf(ac[i], gals[i]), gags[i]);
  }
  for (int o = 16; o; o >>= 1) s += __shfl_xor(s, o);
  if (lane == 0) mr[b] = s * (1.f / 512.f);
  __syncthreads();
  if (tid == 0) {
    float mx = -1e30f;
    for (int i = 0; i < 16; i++) mx = fmaxf(mx, mr[i]);
    float e[16], sum = 0.f;
    for (int i = 0; i < 16; i++) { e[i] = __expf(mr[i] - mx); sum += e[i]; }
    float inv = 1.f / sum;
    for (int i = 0; i < 16; i++) sm[i] = e[i] * inv;
  }
}

// out = sm[b] * x  (vectorized float4; 2^19 float4 per batch)
__global__ void k_scale(const float4* __restrict__ x, const float* __restrict__ sm,
                        float4* __restrict__ out) {
  size_t g = (size_t)blockIdx.x * 256 + threadIdx.x;
  int b = (int)(g >> 19);
  float s = sm[b];
  float4 v = x[g];
  out[g] = make_float4(v.x * s, v.y * s, v.z * s, v.w * s);
}

// ---------------------------------------------------------------------------

extern "C" void kernel_launch(void* const* d_in, const int* in_sizes, int n_in,
                              void* d_out, int out_size, void* d_ws, size_t ws_size,
                              hipStream_t stream) {
  const float* x        = (const float*)d_in[0];
  const float* ac_w0    = (const float*)d_in[1];
  const float* ac_b0    = (const float*)d_in[2];
  const float* ac_w1    = (const float*)d_in[3];
  const float* ac_b1    = (const float*)d_in[4];
  const float* ac_gamma = (const float*)d_in[5];
  const float* ac_beta  = (const float*)d_in[6];
  const float* als_w1   = (const float*)d_in[7];
  const float* als_b1   = (const float*)d_in[8];
  const float* als_w3   = (const float*)d_in[9];
  const float* als_b3   = (const float*)d_in[10];
  const float* als_w2   = (const float*)d_in[11];
  const float* als_b2   = (const float*)d_in[12];
  const float* als_g    = (const float*)d_in[13];
  const float* als_b    = (const float*)d_in[14];
  const float* phi_w    = (const float*)d_in[15];
  const float* theta_w  = (const float*)d_in[16];
  const float* g_w      = (const float*)d_in[17];
  const float* mask_w   = (const float*)d_in[18];

  char* wp = (char*)d_ws;
  auto bump = [&](size_t bytes) -> void* {
    void* p = wp;
    wp += (bytes + 255) & ~(size_t)255;
    return p;
  };
  float*  ap    = (float*)bump(BB * CCH * 4);
  float*  acout = (float*)bump(BB * CCH * 4);
  float*  gals  = (float*)bump(BB * CCH * 4);
  float*  gags  = (float*)bump(BB * CCH * 4);
  float*  s12   = (float*)bump(2 * BB * CCH * 4);
  float*  s1 = s12, *s2 = s12 + BB * CCH;
  float*  om    = (float*)bump(BB * 64 * 4);
  float*  smv   = (float*)bump(64 * 4);
  float*  y1    = (float*)bump((size_t)BB * 32 * HW * 4);
  float*  y3    = (float*)bump((size_t)BB * 32 * HW * 4);
  float*  AttnT = (float*)bump((size_t)BB * 512 * 512 * 4);
  __bf16* Pbf   = (__bf16*)bump((size_t)BB * 64 * HW * 2);
  __bf16* Tbf   = (__bf16*)bump((size_t)BB * 64 * HW * 2);
  __bf16* Gbf   = (__bf16*)bump((size_t)BB * 64 * HW * 2);
  __bf16* At    = (__bf16*)y3;   // alias: y3 dead before softmax runs
  __bf16* Outbf = (__bf16*)y1;   // alias: y1 dead before out-GEMM runs

  const size_t vb = (size_t)64 * HW;  // 262144 = per-batch 512x512 view

  // 1) channel means of x
  k_row_mean_f32<<<BB * CCH, 256, 0, stream>>>(x, ap, HW);
  // 2) AC branch
  k_ac<<<1, 512, 0, stream>>>(ap, ac_w0, ac_b0, ac_w1, ac_b1, ac_gamma, ac_beta, acout);
  // 3) ALS conv1x1 512->32 (WMMA)
  k_conv1x1_wmma<false><<<dim3(32, 1, BB), 256, 0, stream>>>(als_w1, x, als_b1,
                                                             y1, nullptr, 32, CCH, HW);
  // 4) ALS 3x3 conv 32->32
  k_conv3x3<<<dim3(16, 32, BB), 256, 0, stream>>>(y1, als_w3, als_b3, y3);
  // 5/6) ALS conv1x1 32->512 + relu, reduced to sums on the fly
  hipMemsetAsync(s12, 0, 2 * BB * CCH * 4, stream);
  k_als2_reduce<<<dim3(16, BB), 256, 0, stream>>>(y3, als_w2, als_b2, s1, s2);
  // 7) BN-of-mean
  k_gap_als<<<1, 512, 0, stream>>>(s1, s2, als_g, als_b, gals);
  // 8) attention projections phi/theta/g (WMMA, bf16 out)
  k_conv1x1_wmma<true><<<dim3(32, 2, BB), 256, 0, stream>>>(phi_w,   x, nullptr, nullptr, Pbf, 64, CCH, HW);
  k_conv1x1_wmma<true><<<dim3(32, 2, BB), 256, 0, stream>>>(theta_w, x, nullptr, nullptr, Tbf, 64, CCH, HW);
  k_conv1x1_wmma<true><<<dim3(32, 2, BB), 256, 0, stream>>>(g_w,     x, nullptr, nullptr, Gbf, 64, CCH, HW);
  // 9) AttnT[m,n] = sum_c Phi[c,m] * Theta[c,n]  (softmax axis -> rows)
  k_gemm_bf16<true, false><<<dim3(4, 16, BB), 256, 0, stream>>>(
      Pbf, 512, vb, Tbf, 512, vb, AttnT, nullptr, 512, vb, 512);
  // 10) row softmax -> At (bf16), already in B-operand layout for G @ A^T
  k_softmax_rows<<<dim3(512, BB), 256, 0, stream>>>(AttnT, At);
  // 11) Out[c2,n] = G[c2,:] @ At[:,n]
  k_gemm_bf16<false, true><<<dim3(4, 16, BB), 256, 0, stream>>>(
      Gbf, 512, vb, At, 512, vb, nullptr, Outbf, 512, vb, 512);
  // 12) spatial means of Out (per b, ci)
  k_row_mean_bf16<<<BB * 64, 256, 0, stream>>>(Outbf, om, HW);
  // 13) gap_ags = mask_w @ out_mean + mean(x)
  k_gap_ags<<<1, 512, 0, stream>>>(ap, mask_w, om, gags);
  // 14) m = max3, channel-mean, batch softmax
  k_final_sm<<<1, 512, 0, stream>>>(acout, gals, gags, smv);
  // 15) out = sm[b] * x
  k_scale<<<(BB * CCH * HW) / (256 * 4), 256, 0, stream>>>((const float4*)x, smv,
                                                           (float4*)d_out);
}
